// MultiHeadAttentionQuantum_65481071409822
// MI455X (gfx1250) — matrix operand order repaired and checked by reading, not compiled, not run on hardware
//
#include <hip/hip_runtime.h>
#include <stdint.h>

typedef __attribute__((ext_vector_type(16))) _Float16 v16h;
typedef __attribute__((ext_vector_type(4)))  _Float16 v4h;
typedef __attribute__((ext_vector_type(8)))  float    v8f;
typedef __attribute__((ext_vector_type(4)))  unsigned int v4u;

#define EDIM 512
#define NWIRES 8
#define HEADS 64
#define NTOK 65536          // B*S = 16*4096
#define KT 32               // K step per WMMA
#define MT 128              // macrotile M
#define NT 256              // macrotile N
#define AP 40               // LDS row pitch in halves (80B, 16B aligned, skewed)
#define NKSTEP (EDIM / KT)  // 16

// ---- CDNA5 async global<->LDS copies (ASYNCcnt-tracked, bypass VGPRs) ------
__device__ __forceinline__ void async_load_b128(uint32_t lds_off, uint64_t gaddr) {
    asm volatile("global_load_async_to_lds_b128 %0, %1, off"
                 :: "v"(lds_off), "v"(gaddr) : "memory");
}
__device__ __forceinline__ void async_store_b128(uint64_t gaddr, uint32_t lds_off) {
    asm volatile("global_store_async_from_lds_b128 %0, %1, off"
                 :: "v"(gaddr), "v"(lds_off) : "memory");
}
__device__ __forceinline__ void wait_async0() {
    asm volatile("s_wait_asynccnt 0x0" ::: "memory");
}
__device__ __forceinline__ uint32_t lds_addr32(const void* p) {
    return (uint32_t)(uintptr_t)p;   // LDS aperture: addr[31:0] = LDS offset
}

// ---------------------------------------------------------------- fp32 -> fp16
__global__ __launch_bounds__(256)
void cvt_f32_to_f16(const float4* __restrict__ s, v4h* __restrict__ d, int n4) {
    int i = blockIdx.x * blockDim.x + threadIdx.x;
    if (i < n4) {
        float4 f = s[i];
        v4h h = { (_Float16)f.x, (_Float16)f.y, (_Float16)f.z, (_Float16)f.w };
        d[i] = h;
    }
}

// ---------------------------------------------------------------- fused QKV GEMM
// Y = cos(X @ W^T + b + theta[n%8]) for q,k,v, f16 output.
// grid: (6, 512); block 256 (8 wave32, 2x4 wave grid, 64x64 C per wave).
__global__ __launch_bounds__(256)
void qkv_gemm(const _Float16* __restrict__ Xh,
              const _Float16* __restrict__ Wqh, const _Float16* __restrict__ Wkh,
              const _Float16* __restrict__ Wvh,
              const float* __restrict__ bq, const float* __restrict__ bk,
              const float* __restrict__ bv, const float* __restrict__ theta,
              _Float16* __restrict__ Qh, _Float16* __restrict__ Kh,
              _Float16* __restrict__ Vh)
{
    __shared__ alignas(16) union {
        struct { _Float16 A[2][MT][AP]; _Float16 B[2][NT][AP]; } ab;  // 60 KB
        _Float16 c[MT][NT + 8];                                       // 66 KB
    } sm;

    const int nb   = blockIdx.x;           // 0..5
    const int proj = nb >> 1;              // 0=q 1=k 2=v
    const int nOff = (nb & 1) * NT;        // col base within projection
    const int mBase = blockIdx.y * MT;

    const _Float16* __restrict__ Wh   = (proj == 0) ? Wqh : (proj == 1) ? Wkh : Wvh;
    const float*    __restrict__ bias = (proj == 0) ? bq : (proj == 1) ? bk : bv;
    _Float16*       __restrict__ Out  = (proj == 0) ? Qh : (proj == 1) ? Kh : Vh;

    const int tid  = threadIdx.x;
    const int lane = tid & 31;
    const int wid  = tid >> 5;
    const int wm   = (wid & 1) * 64;       // wave M offset
    const int wn   = (wid >> 1) * 64;      // wave N offset
    const int l16  = lane & 15;
    const int hi   = lane >> 4;

    auto fill = [&](int buf, int k0) {
        // A: 128 rows x 32 halves; thread copies 32B of row tid>>1
        const _Float16* ga = Xh + (size_t)(mBase + (tid >> 1)) * EDIM + k0 + (tid & 1) * 16;
        const uint32_t la = lds_addr32(&sm.ab.A[buf][tid >> 1][(tid & 1) * 16]);
        async_load_b128(la,      (uint64_t)(uintptr_t)ga);
        async_load_b128(la + 16, (uint64_t)(uintptr_t)ga + 16);
        // B: 256 rows x 32 halves; thread copies row tid (64B)
        const _Float16* gb = Wh + (size_t)(nOff + tid) * EDIM + k0;
        const uint32_t lb = lds_addr32(&sm.ab.B[buf][tid][0]);
        async_load_b128(lb,      (uint64_t)(uintptr_t)gb);
        async_load_b128(lb + 16, (uint64_t)(uintptr_t)gb + 16);
        async_load_b128(lb + 32, (uint64_t)(uintptr_t)gb + 32);
        async_load_b128(lb + 48, (uint64_t)(uintptr_t)gb + 48);
    };

    v8f zero = {0.f,0.f,0.f,0.f,0.f,0.f,0.f,0.f};
    v8f acc[4][4];
#pragma unroll
    for (int a = 0; a < 4; ++a)
#pragma unroll
        for (int c = 0; c < 4; ++c) acc[a][c] = zero;

    fill(0, 0);
    for (int k = 0; k < NKSTEP; ++k) {
        const int buf = k & 1;
        wait_async0();             // own fills for step k complete
        __syncthreads();           // all fills visible; prior reads of buf done
        if (k + 1 < NKSTEP) fill(buf ^ 1, (k + 1) * KT);

        union { v16h h; v4u u[2]; } bf[4];
#pragma unroll
        for (int tn = 0; tn < 4; ++tn) {
            const _Float16* p = &sm.ab.B[buf][wn + tn * 16 + l16][hi * 8];
            bf[tn].u[0] = *(const v4u*)p;
            bf[tn].u[1] = *(const v4u*)(p + 16);
        }
#pragma unroll
        for (int tm = 0; tm < 4; ++tm) {
            union { v16h h; v4u u[2]; } af;
            const _Float16* p = &sm.ab.A[buf][wm + tm * 16 + l16][hi * 8];
            af.u[0] = *(const v4u*)p;
            af.u[1] = *(const v4u*)(p + 16);
#pragma unroll
            for (int tn = 0; tn < 4; ++tn)
                acc[tm][tn] = __builtin_amdgcn_wmma_f32_16x16x32_f16(
                    false, af.h, false, bf[tn].h, (short)0, acc[tm][tn], false, false);
        }
        __syncthreads();           // reads of buf consumed before refill at k+1
    }

    // epilogue: cos(acc + bias + theta) -> LDS tile -> async LDS->global stores
    const int rsel = hi * 8;
#pragma unroll
    for (int tm = 0; tm < 4; ++tm)
#pragma unroll
        for (int tn = 0; tn < 4; ++tn) {
            const int n = nOff + wn + tn * 16 + l16;       // 0..511 within proj
            const float bth = bias[n] + theta[n & 7];
#pragma unroll
            for (int j = 0; j < 8; ++j)
                sm.c[wm + tm * 16 + rsel + j][wn + tn * 16 + l16] =
                    (_Float16)__cosf(acc[tm][tn][j] + bth);
        }
    __syncthreads();
    {
        const int srow = tid >> 1;
        const int scol = (tid & 1) * 128;  // halves
        _Float16* dst = Out + (size_t)(mBase + srow) * EDIM + nOff + scol;
        const uint32_t lsrc = lds_addr32(&sm.c[srow][scol]);
#pragma unroll
        for (int u = 0; u < 16; ++u)
            async_store_b128((uint64_t)(uintptr_t)(dst + u * 8), lsrc + u * 16);
        wait_async0();
    }
}

// ---------------------------------------------------------------- per-token attention
// 64 threads per token (4 tokens/block). Softmax across heads (64x64, dk=8).
// Writes the reference's scrambled [B,H,S,dk]->[B,4096,512] layout directly.
__global__ __launch_bounds__(256)
void attn_kernel(const _Float16* __restrict__ Qh, const _Float16* __restrict__ Kh,
                 const _Float16* __restrict__ Vh, _Float16* __restrict__ Ah)
{
    __shared__ float kf[4][HEADS][NWIRES];   // 8 KB
    __shared__ float vf[4][HEADS][NWIRES];   // 8 KB

    const int tid = threadIdx.x;
    const int tl  = tid >> 6;                // token within block
    const int i   = tid & 63;                // head / score row
    const size_t token = (size_t)blockIdx.x * 4 + tl;
    const size_t base  = token * EDIM;

    float q[NWIRES];
    {
        union { v4u u; _Float16 h[8]; } t;
        t.u = *(const v4u*)(Qh + base + i * 8);
#pragma unroll
        for (int d = 0; d < 8; ++d) q[d] = (float)t.h[d];
        t.u = *(const v4u*)(Kh + base + i * 8);
#pragma unroll
        for (int d = 0; d < 8; ++d) kf[tl][i][d] = (float)t.h[d];
        t.u = *(const v4u*)(Vh + base + i * 8);
#pragma unroll
        for (int d = 0; d < 8; ++d) vf[tl][i][d] = (float)t.h[d];
    }
    __syncthreads();

    const float scale = 0.35355339059327373f;  // 1/sqrt(8)
    float s[HEADS];
    float mx = -3.0e38f;
#pragma unroll
    for (int j = 0; j < HEADS; ++j) {
        float a = 0.f;
#pragma unroll
        for (int d = 0; d < 8; ++d) a = __builtin_fmaf(q[d], kf[tl][j][d], a);
        a *= scale;
        s[j] = a;
        mx = fmaxf(mx, a);
    }
    float sum = 0.f;
#pragma unroll
    for (int j = 0; j < HEADS; ++j) { s[j] = __expf(s[j] - mx); sum += s[j]; }
    const float inv = 1.0f / sum;

    float o[NWIRES] = {0.f,0.f,0.f,0.f,0.f,0.f,0.f,0.f};
#pragma unroll
    for (int j = 0; j < HEADS; ++j) {
        const float p = s[j];
#pragma unroll
        for (int d = 0; d < 8; ++d) o[d] = __builtin_fmaf(p, vf[tl][j][d], o[d]);
    }

    // scrambled destination: row = b*4096 + h*64 + s/64, col = (s%64)*8 + d
    const int b    = (int)(token >> 12);
    const int sIdx = (int)(token & 4095);
    const size_t row = (size_t)b * 4096 + (size_t)i * 64 + (sIdx >> 6);
    const size_t col = (size_t)(sIdx & 63) * 8;
    union { v4u u; _Float16 h[8]; } w;
#pragma unroll
    for (int d = 0; d < 8; ++d) w.h[d] = (_Float16)(o[d] * inv);
    *(v4u*)(Ah + row * EDIM + col) = w.u;
}

// ---------------------------------------------------------------- output GEMM
// out = attn @ Wo^T + bo, fp32 output. grid: (2, 512), block 256.
__global__ __launch_bounds__(256)
void out_gemm(const _Float16* __restrict__ Ahm, const _Float16* __restrict__ Woh,
              const float* __restrict__ bo, float* __restrict__ out)
{
    __shared__ alignas(16) union {
        struct { _Float16 A[2][MT][AP]; _Float16 B[2][NT][AP]; } ab;  // 60 KB
        float c[MT][NT + 4];                                          // 130 KB
    } sm;

    const int nOff  = blockIdx.x * NT;
    const int mBase = blockIdx.y * MT;

    const int tid  = threadIdx.x;
    const int lane = tid & 31;
    const int wid  = tid >> 5;
    const int wm   = (wid & 1) * 64;
    const int wn   = (wid >> 1) * 64;
    const int l16  = lane & 15;
    const int hi   = lane >> 4;

    auto fill = [&](int buf, int k0) {
        const _Float16* ga = Ahm + (size_t)(mBase + (tid >> 1)) * EDIM + k0 + (tid & 1) * 16;
        const uint32_t la = lds_addr32(&sm.ab.A[buf][tid >> 1][(tid & 1) * 16]);
        async_load_b128(la,      (uint64_t)(uintptr_t)ga);
        async_load_b128(la + 16, (uint64_t)(uintptr_t)ga + 16);
        const _Float16* gb = Woh + (size_t)(nOff + tid) * EDIM + k0;
        const uint32_t lb = lds_addr32(&sm.ab.B[buf][tid][0]);
        async_load_b128(lb,      (uint64_t)(uintptr_t)gb);
        async_load_b128(lb + 16, (uint64_t)(uintptr_t)gb + 16);
        async_load_b128(lb + 32, (uint64_t)(uintptr_t)gb + 32);
        async_load_b128(lb + 48, (uint64_t)(uintptr_t)gb + 48);
    };

    v8f zero = {0.f,0.f,0.f,0.f,0.f,0.f,0.f,0.f};
    v8f acc[4][4];
#pragma unroll
    for (int a = 0; a < 4; ++a)
#pragma unroll
        for (int c = 0; c < 4; ++c) acc[a][c] = zero;

    fill(0, 0);
    for (int k = 0; k < NKSTEP; ++k) {
        const int buf = k & 1;
        wait_async0();
        __syncthreads();
        if (k + 1 < NKSTEP) fill(buf ^ 1, (k + 1) * KT);

        union { v16h h; v4u u[2]; } bf[4];
#pragma unroll
        for (int tn = 0; tn < 4; ++tn) {
            const _Float16* p = &sm.ab.B[buf][wn + tn * 16 + l16][hi * 8];
            bf[tn].u[0] = *(const v4u*)p;
            bf[tn].u[1] = *(const v4u*)(p + 16);
        }
#pragma unroll
        for (int tm = 0; tm < 4; ++tm) {
            union { v16h h; v4u u[2]; } af;
            const _Float16* p = &sm.ab.A[buf][wm + tm * 16 + l16][hi * 8];
            af.u[0] = *(const v4u*)p;
            af.u[1] = *(const v4u*)(p + 16);
#pragma unroll
            for (int tn = 0; tn < 4; ++tn)
                acc[tm][tn] = __builtin_amdgcn_wmma_f32_16x16x32_f16(
                    false, af.h, false, bf[tn].h, (short)0, acc[tm][tn], false, false);
        }
        __syncthreads();
    }

    // epilogue: acc + bias -> LDS (f32) -> async LDS->global stores
    const int rsel = hi * 8;
#pragma unroll
    for (int tm = 0; tm < 4; ++tm)
#pragma unroll
        for (int tn = 0; tn < 4; ++tn) {
            const int n = nOff + wn + tn * 16 + l16;
            const float bb = bo[n];
#pragma unroll
            for (int j = 0; j < 8; ++j)
                sm.c[wm + tm * 16 + rsel + j][wn + tn * 16 + l16] = acc[tm][tn][j] + bb;
        }
    __syncthreads();
    {
        const int srow = tid >> 1;
        const int scol = (tid & 1) * 128;  // floats
        float* dst = out + (size_t)(mBase + srow) * EDIM + nOff + scol;
        const uint32_t lsrc = lds_addr32(&sm.c[srow][scol]);
#pragma unroll
        for (int u = 0; u < 32; ++u)
            async_store_b128((uint64_t)(uintptr_t)(dst + u * 4), lsrc + u * 16);
        wait_async0();
    }
}

// ---------------------------------------------------------------- launch
extern "C" void kernel_launch(void* const* d_in, const int* in_sizes, int n_in,
                              void* d_out, int out_size, void* d_ws, size_t ws_size,
                              hipStream_t stream) {
    const float* x     = (const float*)d_in[0];
    const float* Wq    = (const float*)d_in[1];
    const float* bq    = (const float*)d_in[2];
    const float* Wk    = (const float*)d_in[3];
    const float* bk    = (const float*)d_in[4];
    const float* Wv    = (const float*)d_in[5];
    const float* bv    = (const float*)d_in[6];
    const float* Wo    = (const float*)d_in[7];
    const float* bo    = (const float*)d_in[8];
    const float* theta = (const float*)d_in[9];

    const size_t nX = (size_t)NTOK * EDIM;      // 33,554,432
    const size_t nW = (size_t)EDIM * EDIM;      // 262,144

    char* ws = (char*)d_ws;
    _Float16* xh  = (_Float16*)ws;              // 67 MB; reused as attn matrix
    _Float16* wqh = (_Float16*)(ws + nX * 2);
    _Float16* wkh = wqh + nW;
    _Float16* wvh = wkh + nW;
    _Float16* woh = wvh + nW;
    _Float16* qh  = (_Float16*)((char*)(woh + nW));
    _Float16* kh  = qh + nX;
    _Float16* vh  = kh + nX;

    // fp32 -> f16 staging (x fits L2 afterwards: 67 MB < 192 MB)
    cvt_f32_to_f16<<<(int)(nX / 4 / 256), 256, 0, stream>>>((const float4*)x,  (v4h*)xh,  (int)(nX / 4));
    cvt_f32_to_f16<<<(int)(nW / 4 / 256), 256, 0, stream>>>((const float4*)Wq, (v4h*)wqh, (int)(nW / 4));
    cvt_f32_to_f16<<<(int)(nW / 4 / 256), 256, 0, stream>>>((const float4*)Wk, (v4h*)wkh, (int)(nW / 4));
    cvt_f32_to_f16<<<(int)(nW / 4 / 256), 256, 0, stream>>>((const float4*)Wv, (v4h*)wvh, (int)(nW / 4));
    cvt_f32_to_f16<<<(int)(nW / 4 / 256), 256, 0, stream>>>((const float4*)Wo, (v4h*)woh, (int)(nW / 4));

    // fused QKV projection + cos epilogue (WMMA f16 -> f32 acc, async LDS staging)
    qkv_gemm<<<dim3(6, NTOK / MT), 256, 0, stream>>>(xh, wqh, wkh, wvh,
                                                     bq, bk, bv, theta, qh, kh, vh);

    // per-token cross-head attention; writes scrambled layout into xh (x dead now)
    attn_kernel<<<NTOK / 4, 256, 0, stream>>>(qh, kh, vh, xh);

    // final projection, fp32 out
    out_gemm<<<dim3(EDIM / NT, NTOK / MT), 256, 0, stream>>>(xh, woh, bo, (float*)d_out);
}
